// RNN_77343771066978
// MI455X (gfx1250) — matrix operand order repaired and checked by reading, not compiled, not run on hardware
//
#include <hip/hip_runtime.h>

#define SEQ   4096
#define BATCH 64
#define INF_  64          // input features
#define HID   256
#define G3    768         // 3*HID

#define GAS __attribute__((address_space(1)))   // global address space

typedef __attribute__((ext_vector_type(16))) __bf16 v16bf;
typedef __attribute__((ext_vector_type(8)))  __bf16 v8bf;
typedef __attribute__((ext_vector_type(8)))  float  v8f;

union V16U { v16bf v; v8bf h[2]; };

// ---------------- LDS layout (bytes) ----------------
#define HSTR   264                                  // 256 + 8 pad elems per row
#define XSTR   72                                   // 64 + 8 pad
#define OFF_HF 0                                    // h as f32      [64][HSTR]
#define OFF_HB (OFF_HF + 64*HSTR*4)                 // h as bf16     [64][HSTR]
#define OFF_XB (OFF_HB + 64*HSTR*2)                 // x_t as bf16   [64][XSTR]
#define OFF_RB (OFF_XB + 64*XSTR*2)                 // r gate bf16   [64][HSTR]
#define OFF_ZB (OFF_RB + 64*HSTR*2)                 // z gate bf16   [64][HSTR]
#define OFF_OS (OFF_ZB + 64*HSTR*2)                 // out accum f32 [64]
#define SMEM_BYTES (OFF_OS + 64*4)                  // = 178432 B (< 320 KB/WGP)

// A-tile (bf16 16x32, ISA 7.12.2): lane holds row m0+(lane&15);
// chunk0 = k0 + 8*half .. +7, chunk1 = k0 + 16 + 8*half .. +7 (two 16B ds loads)
__device__ __forceinline__ v16bf load_a_lds(const __bf16* base, int stride,
                                            int row, int k0, int half) {
  V16U u;
  const __bf16* p = base + row * stride + k0 + 8 * half;
  u.h[0] = *(const v8bf*)(p);
  u.h[1] = *(const v8bf*)(p + 16);
  return u.v;
}

// B-tile (bf16 32x16): lane holds column gcol; chunk0 = k0 + 8*half .. +7,
// chunk1 = k0 + 16 + 8*half .. +7 of the (row-major [G][K]) weight row, via true
// global_load_b128 (address-space 1, SGPR base + 32-bit offset).
__device__ __forceinline__ v16bf load_b_glb(const GAS __bf16* w, int ldk,
                                            int gcol, int k0, int half) {
  V16U u;
  const GAS __bf16* p = w + (gcol * ldk + k0 + 8 * half);
  u.h[0] = *(const GAS v8bf*)(p);
  u.h[1] = *(const GAS v8bf*)(p + 16);
  return u.v;
}

__device__ __forceinline__ v8f wmma_bf(v16bf a, v16bf b, v8f c) {
  // (neg_a, A, neg_b, B, c_mod, C, reuse_a, reuse_b)
  return __builtin_amdgcn_wmma_f32_16x16x32_bf16(false, a, false, b, (short)0, c,
                                                 false, false);
}

__device__ __forceinline__ float sigmoidf_(float x) { return 1.f / (1.f + __expf(-x)); }
__device__ __forceinline__ float tanhf_(float x) {
  x = fminf(fmaxf(x, -15.f), 15.f);
  float e = __expf(2.f * x);
  return (e - 1.f) / (e + 1.f);
}

// ---------------- pre-pass: f32 weights -> bf16 in workspace ----------------
__global__ void cvt_weights(const float* __restrict__ w_ih, const float* __restrict__ w_hh,
                            __bf16* __restrict__ wih_bf, __bf16* __restrict__ whh_bf) {
  int i = blockIdx.x * blockDim.x + threadIdx.x;
  if (i < G3 * INF_) wih_bf[i] = (__bf16)w_ih[i];
  if (i < G3 * HID)  whh_bf[i] = (__bf16)w_hh[i];
}

// ---------------- sequential GRU: one workgroup, 32 waves ----------------
__global__ __launch_bounds__(1024, 1)
void gru_seq_kernel(const float* __restrict__ x,
                    const __bf16* __restrict__ wih_bf,
                    const __bf16* __restrict__ whh_bf,
                    const float* __restrict__ b_ih,
                    const float* __restrict__ b_hh,
                    const float* __restrict__ fc_w,
                    const float* __restrict__ fc_b,
                    float* __restrict__ out) {
  extern __shared__ char smem[];
  float*  h_f32  = (float*)(smem + OFF_HF);
  __bf16* h_bf   = (__bf16*)(smem + OFF_HB);
  __bf16* x_bf   = (__bf16*)(smem + OFF_XB);
  __bf16* r_bf   = (__bf16*)(smem + OFF_RB);
  __bf16* z_bf   = (__bf16*)(smem + OFF_ZB);
  float*  out_sm = (float*)(smem + OFF_OS);

  const int tid  = threadIdx.x;
  const int lane = tid & 31;
  const int wave = tid >> 5;     // 0..31
  const int half = lane >> 4;
  const int ln   = lane & 15;

  for (int i = tid; i < 64 * HSTR; i += 1024) { h_f32[i] = 0.f; h_bf[i] = (__bf16)0.f; }
  __syncthreads();

  const float fcb = fc_b[0];

  // ---- per-wave constants (cheap, OK to hoist) ----
  // phase 1: wave == nt (0..31), all 4 M-tiles; gate cols g1 = wave*16 + ln
  const int g1 = wave * 16 + ln;
  const float bias1 = b_ih[g1] + b_hh[g1];
  // phase 2: nt2 = wave>>1 (0..15), 2 M-tiles starting at mtb2
  const int nt2 = wave >> 1;
  const int mtb2 = (wave & 1) * 2;
  const int hc2 = nt2 * 16 + ln;          // column within HID
  const int g2 = 2 * HID + hc2;           // n-gate weight row, 512..767
  const float biasX = b_ih[g2];
  const float biasH = b_hh[g2];
  const float fw2 = fc_w[hc2];

  for (int s = 0; s < SEQ; ++s) {
    // Launder weight bases so LICM cannot hoist the (loop-invariant) B-tile
    // loads out of the s-loop and spill them to scratch. Cast the laundered
    // address into address-space(1) so loads lower to global_load_b128
    // (LOADcnt only), not flat_load (which also ties up DScnt / LDS path).
    unsigned long long whh_u = (unsigned long long)whh_bf;
    unsigned long long wih_u = (unsigned long long)wih_bf;
    asm volatile("" : "+s"(whh_u), "+s"(wih_u));
    const GAS __bf16* whh_p = (const GAS __bf16*)whh_u;
    const GAS __bf16* wih_p = (const GAS __bf16*)wih_u;

    // ---- A: stage x_t (f32 -> bf16 LDS), zero fc accumulator ----
    if (tid < 64) out_sm[tid] = 0.f;
    {
      const float4* x4 = (const float4*)(x + (size_t)s * (BATCH * INF_));
      float4 f = x4[tid];
      int e0 = tid * 4;
      int row = e0 >> 6, col = e0 & 63;
      __bf16* p = &x_bf[row * XSTR + col];
      p[0] = (__bf16)f.x; p[1] = (__bf16)f.y; p[2] = (__bf16)f.z; p[3] = (__bf16)f.w;
      if (s + 1 < SEQ) __builtin_prefetch(x + (size_t)(s + 1) * (BATCH * INF_) + e0, 0, 1);
    }
    __syncthreads();  // S1: x ready

    // ---- B: phase 1 -- r and z gates. One N-group per wave, B reused x4. ----
    {
      v8f acc0, acc1, acc2, acc3;
      #pragma unroll
      for (int i = 0; i < 8; ++i) { acc0[i] = bias1; acc1[i] = bias1; acc2[i] = bias1; acc3[i] = bias1; }
      #pragma unroll
      for (int k0 = 0; k0 < HID; k0 += 32) {
        const v16bf a0 = load_a_lds(h_bf, HSTR,  0 + ln, k0, half);
        const v16bf a1 = load_a_lds(h_bf, HSTR, 16 + ln, k0, half);
        const v16bf a2 = load_a_lds(h_bf, HSTR, 32 + ln, k0, half);
        const v16bf a3 = load_a_lds(h_bf, HSTR, 48 + ln, k0, half);
        const v16bf b  = load_b_glb(whh_p, HID, g1, k0, half);
        acc0 = wmma_bf(a0, b, acc0);
        acc1 = wmma_bf(a1, b, acc1);
        acc2 = wmma_bf(a2, b, acc2);
        acc3 = wmma_bf(a3, b, acc3);
      }
      #pragma unroll
      for (int k0 = 0; k0 < INF_; k0 += 32) {
        const v16bf a0 = load_a_lds(x_bf, XSTR,  0 + ln, k0, half);
        const v16bf a1 = load_a_lds(x_bf, XSTR, 16 + ln, k0, half);
        const v16bf a2 = load_a_lds(x_bf, XSTR, 32 + ln, k0, half);
        const v16bf a3 = load_a_lds(x_bf, XSTR, 48 + ln, k0, half);
        const v16bf b  = load_b_glb(wih_p, INF_, g1, k0, half);
        acc0 = wmma_bf(a0, b, acc0);
        acc1 = wmma_bf(a1, b, acc1);
        acc2 = wmma_bf(a2, b, acc2);
        acc3 = wmma_bf(a3, b, acc3);
      }
      __bf16* dst = (wave < 16) ? (r_bf + g1) : (z_bf + (g1 - 256));
      #pragma unroll
      for (int i = 0; i < 8; ++i) {
        const int mrow = half * 8 + i;     // C/D row layout (ISA 7.12.2)
        dst[(mrow +  0) * HSTR] = (__bf16)sigmoidf_(acc0[i]);
        dst[(mrow + 16) * HSTR] = (__bf16)sigmoidf_(acc1[i]);
        dst[(mrow + 32) * HSTR] = (__bf16)sigmoidf_(acc2[i]);
        dst[(mrow + 48) * HSTR] = (__bf16)sigmoidf_(acc3[i]);
      }
    }
    __syncthreads();  // S2: r,z ready

    // ---- C: phase 2 -- n gate + candidate h_new (B reused x2) ----
    float hn[2][8];
    {
      v8f aX0, aX1, aH0, aH1;
      #pragma unroll
      for (int i = 0; i < 8; ++i) { aX0[i] = biasX; aX1[i] = biasX; aH0[i] = biasH; aH1[i] = biasH; }
      #pragma unroll
      for (int k0 = 0; k0 < HID; k0 += 32) {
        const v16bf a0 = load_a_lds(h_bf, HSTR, (mtb2 + 0) * 16 + ln, k0, half);
        const v16bf a1 = load_a_lds(h_bf, HSTR, (mtb2 + 1) * 16 + ln, k0, half);
        const v16bf b  = load_b_glb(whh_p, HID, g2, k0, half);
        aH0 = wmma_bf(a0, b, aH0);
        aH1 = wmma_bf(a1, b, aH1);
      }
      #pragma unroll
      for (int k0 = 0; k0 < INF_; k0 += 32) {
        const v16bf a0 = load_a_lds(x_bf, XSTR, (mtb2 + 0) * 16 + ln, k0, half);
        const v16bf a1 = load_a_lds(x_bf, XSTR, (mtb2 + 1) * 16 + ln, k0, half);
        const v16bf b  = load_b_glb(wih_p, INF_, g2, k0, half);
        aX0 = wmma_bf(a0, b, aX0);
        aX1 = wmma_bf(a1, b, aX1);
      }
      #pragma unroll
      for (int j = 0; j < 2; ++j) {
        #pragma unroll
        for (int i = 0; i < 8; ++i) {
          const int m = (mtb2 + j) * 16 + half * 8 + i;
          const float r = (float)r_bf[m * HSTR + hc2];
          const float z = (float)z_bf[m * HSTR + hc2];
          const float preX = (j == 0) ? aX0[i] : aX1[i];
          const float preH = (j == 0) ? aH0[i] : aH1[i];
          const float n = tanhf_(preX + r * preH);  // torch GRU: r gates hidden n-term
          const float hold = h_f32[m * HSTR + hc2];
          hn[j][i] = (1.f - z) * n + z * hold;
        }
      }
    }
    __syncthreads();  // S3: every wave finished reading old h

    // ---- D: commit h_new, fold in fc projection (row-reduce + LDS atomic) ----
    #pragma unroll
    for (int j = 0; j < 2; ++j) {
      #pragma unroll
      for (int i = 0; i < 8; ++i) {
        const int m = (mtb2 + j) * 16 + half * 8 + i;
        const float v = hn[j][i];
        h_f32[m * HSTR + hc2] = v;
        h_bf[m * HSTR + hc2]  = (__bf16)v;
        float p = v * fw2;                 // reduce over the 16 cols of this tile
        p += __shfl_xor(p, 1);
        p += __shfl_xor(p, 2);
        p += __shfl_xor(p, 4);
        p += __shfl_xor(p, 8);
        if (ln == 0) atomicAdd(&out_sm[m], p);   // ds_add_f32
      }
    }
    __syncthreads();  // S4: h and out_sm complete

    // ---- E: emit out[s, :] ----
    if (tid < 64) out[(size_t)s * 64 + tid] = out_sm[tid] + fcb;
  }
}

extern "C" void kernel_launch(void* const* d_in, const int* in_sizes, int n_in,
                              void* d_out, int out_size, void* d_ws, size_t ws_size,
                              hipStream_t stream) {
  (void)in_sizes; (void)n_in; (void)out_size; (void)ws_size;
  const float* x    = (const float*)d_in[0];
  const float* w_ih = (const float*)d_in[1];
  const float* w_hh = (const float*)d_in[2];
  const float* b_ih = (const float*)d_in[3];
  const float* b_hh = (const float*)d_in[4];
  const float* fc_w = (const float*)d_in[5];
  const float* fc_b = (const float*)d_in[6];
  float* out = (float*)d_out;

  __bf16* wih_bf = (__bf16*)d_ws;                                  // 768*64  bf16
  __bf16* whh_bf = (__bf16*)((char*)d_ws + (size_t)G3 * INF_ * 2); // 768*256 bf16

  cvt_weights<<<(G3 * HID + 255) / 256, 256, 0, stream>>>(w_ih, w_hh, wih_bf, whh_bf);
  gru_seq_kernel<<<1, 1024, SMEM_BYTES, stream>>>(x, wih_bf, whh_bf,
                                                  b_ih, b_hh, fc_w, fc_b, out);
}
